// LSTMCore_28475633173192
// MI455X (gfx1250) — compile-verified, matrix operand order checked by reading, not run on hardware
//
#include <hip/hip_runtime.h>

#define USE_TDM 1

typedef __bf16 v8bf  __attribute__((ext_vector_type(8)));
typedef __bf16 v16bf __attribute__((ext_vector_type(16)));
typedef float  v8f   __attribute__((ext_vector_type(8)));
typedef unsigned int v4u __attribute__((ext_vector_type(4)));
typedef int v4i __attribute__((ext_vector_type(4)));
typedef int v8i __attribute__((ext_vector_type(8)));

namespace {
constexpr int kB  = 4096;   // batch
constexpr int kIn = 1024;   // input features (== K for x GEMM)
constexpr int kH  = 1024;   // hidden (== K for h GEMM)
constexpr int kG  = 4096;   // 4*H gate columns (== N)
constexpr int kL  = 8;      // layers/steps

// GEMM tiling: block computes BM x BN, 8 waves in 4(M) x 2(N), wave tile 32x32
constexpr int BM = 128, BN = 64, BK = 32;
constexpr int ROWB   = 72;                 // LDS bytes per 32-elem bf16 row (64B data + 8B pad)
constexpr int ABYTES = BM * ROWB;          // 9216
constexpr int BBYTES = BN * ROWB;          // 4608
constexpr int BUFB   = 2 * ABYTES + 2 * BBYTES; // 27648 per buffer; x2 = 54 KB LDS
}

// ---------- fp32 <-> bf16 split helpers (round-to-nearest-even) ----------
__device__ __forceinline__ unsigned short f2bf(float x) {
  unsigned u = __float_as_uint(x);
  unsigned r = u + 0x7FFFu + ((u >> 16) & 1u);
  return (unsigned short)(r >> 16);
}
__device__ __forceinline__ float bf2f(unsigned short b) {
  return __uint_as_float(((unsigned)b) << 16);
}

__device__ __forceinline__ v16bf ld_frag(const char* p) {
  v8bf a = *(const v8bf*)(p);
  v8bf b = *(const v8bf*)(p + 32);
  return __builtin_shufflevector(a, b, 0,1,2,3,4,5,6,7,8,9,10,11,12,13,14,15);
}

#if USE_TDM
// ---------- CDNA5 Tensor Data Mover: load a rows x 32 bf16 tile (row stride K
// elements) into LDS at ldsAddr, inserting 8B pad after every 64B row. ----------
__device__ __forceinline__ void tdm_load_tile(const unsigned short* g, unsigned ldsAddr,
                                              int rows, int K) {
  unsigned long long ga = (unsigned long long)(size_t)g;
  // D# group0: count=1 | lds_addr | global_addr[56:0] | type=2
  v4u g0 = { 1u,
             ldsAddr,
             (unsigned)ga,
             (unsigned)(ga >> 32) | (2u << 30) };
  // D# group1:
  //  w0: data_size=1 (2B) <<16 | pad_enable<<20 | pad_interval=3 (16 dwords)<<22
  //      | pad_amount=1 (2 dwords)<<25 ; workgroup_mask=0
  //  w1: tensor_dim0[15:0]=32 in [31:16]
  //  w2: tensor_dim1[15:0]=rows in [31:16]
  //  w3: tile_dim0=32 in [31:16]
  //  w4: tile_dim1=rows in [15:0]
  //  w5: tensor_dim0_stride[31:0] = K (elements)
  v8i g1 = { (int)((1u << 16) | (1u << 20) | (3u << 22) | (1u << 25)),
             (int)(32u << 16),
             (int)((unsigned)rows << 16),
             (int)(32u << 16),
             rows,
             K,
             0, 0 };
  v4i z4 = {0, 0, 0, 0};
#if defined(__clang_major__) && (__clang_major__ >= 23)
  v8i z8 = {0, 0, 0, 0, 0, 0, 0, 0};
  __builtin_amdgcn_tensor_load_to_lds(g0, g1, z4, z4, z8, 0);
#else
  __builtin_amdgcn_tensor_load_to_lds(g0, g1, z4, z4, 0);
#endif
}
#else
// ---------- fallback: per-lane async global->LDS staging ----------
__device__ __forceinline__ void async_cp16(unsigned lds_off, const void* g) {
  asm volatile("global_load_async_to_lds_b128 %0, %1, off"
               :: "v"(lds_off), "v"((unsigned long long)(size_t)g)
               : "memory");
}
__device__ __forceinline__ void stage_rowchunk(const unsigned short* __restrict__ src,
                                               unsigned ldsBase, int r0, int k0, int K, int c) {
  int row = c >> 2, seg = c & 3;
  async_cp16(ldsBase + (unsigned)(row * ROWB + seg * 16),
             src + (size_t)(r0 + row) * K + k0 + seg * 8);
}
#endif

// ---------- split fp32 -> (bf16 hi, bf16 lo) ----------
__global__ __launch_bounds__(256) void lstm_split_kernel(
    const float* __restrict__ src, unsigned short* __restrict__ hi,
    unsigned short* __restrict__ lo, int n) {
  int i = blockIdx.x * 256 + threadIdx.x;
  if (i >= n) return;
  float x = src[i];
  unsigned short h = f2bf(x);
  hi[i] = h;
  lo[i] = f2bf(x - bf2f(h));
}

// ---------- NT GEMM: C[M,N] = Arow[M,K] . Brow[N,K]^T + bias[N] ----------
// A,B given as split bf16 hi/lo; fp32 accumulate via 3x wmma_f32_16x16x32_bf16.
__global__ __launch_bounds__(256) void lstm_gemm_nt(
    const unsigned short* __restrict__ Ahi, const unsigned short* __restrict__ Alo,
    const unsigned short* __restrict__ Bhi, const unsigned short* __restrict__ Blo,
    const float* __restrict__ bias, float* __restrict__ C,
    int M, int N, int K) {
  __shared__ char sm[2 * BUFB];

  const int t    = threadIdx.x;
  const int lane = t & 31;
  const int wave = t >> 5;
  // SGPR copy of the wave id: staging selection becomes a scalar branch.
  const int waveu = __builtin_amdgcn_readfirstlane(wave);
  const int wm   = wave >> 1;   // 0..3 : wave M offset = wm*32
  const int wn   = wave & 1;    // 0..1 : wave N offset = wn*32
  const int r    = lane & 15;   // row/col within 16
  const int half = lane >> 4;   // K-half select

  const int m0 = blockIdx.y * BM;
  const int n0 = blockIdx.x * BN;

  const unsigned smBase = (unsigned)(size_t)(&sm[0]);

#if USE_TDM
  auto stage_all = [&](int bufSel, int k0) {
    unsigned base = smBase + (unsigned)(bufSel * BUFB);
    if (waveu == 0)
      tdm_load_tile(Ahi + (size_t)m0 * K + k0, base, BM, K);
    else if (waveu == 1)
      tdm_load_tile(Alo + (size_t)m0 * K + k0, base + ABYTES, BM, K);
    else if (waveu == 2)
      tdm_load_tile(Bhi + (size_t)n0 * K + k0, base + 2 * ABYTES, BN, K);
    else if (waveu == 3)
      tdm_load_tile(Blo + (size_t)n0 * K + k0, base + 2 * ABYTES + BBYTES, BN, K);
  };
#else
  auto stage_all = [&](int bufSel, int k0) {
    unsigned base = smBase + (unsigned)(bufSel * BUFB);
    stage_rowchunk(Ahi, base,                       m0, k0, K, t);
    stage_rowchunk(Ahi, base,                       m0, k0, K, t + 256);
    stage_rowchunk(Alo, base + ABYTES,              m0, k0, K, t);
    stage_rowchunk(Alo, base + ABYTES,              m0, k0, K, t + 256);
    stage_rowchunk(Bhi, base + 2 * ABYTES,          n0, k0, K, t);
    stage_rowchunk(Blo, base + 2 * ABYTES + BBYTES, n0, k0, K, t);
  };
#endif

  v8f acc[2][2] = {};

  auto compute_buf = [&](int cur) {
    const char* buf = sm + cur * BUFB;
    v16bf aH[2], aL[2], bH[2], bL[2];
#pragma unroll
    for (int mt = 0; mt < 2; ++mt) {
      unsigned off = (unsigned)((wm * 32 + mt * 16 + r) * ROWB + half * 16);
      aH[mt] = ld_frag(buf + off);
      aL[mt] = ld_frag(buf + off + ABYTES);
    }
#pragma unroll
    for (int nt = 0; nt < 2; ++nt) {
      unsigned off = (unsigned)(2 * ABYTES + (wn * 32 + nt * 16 + r) * ROWB + half * 16);
      bH[nt] = ld_frag(buf + off);
      bL[nt] = ld_frag(buf + off + BBYTES);
    }
#pragma unroll
    for (int mt = 0; mt < 2; ++mt)
#pragma unroll
      for (int nt = 0; nt < 2; ++nt) {
        acc[mt][nt] = __builtin_amdgcn_wmma_f32_16x16x32_bf16(
            false, aH[mt], false, bH[nt], (short)0, acc[mt][nt], false, false);
        acc[mt][nt] = __builtin_amdgcn_wmma_f32_16x16x32_bf16(
            false, aH[mt], false, bL[nt], (short)0, acc[mt][nt], false, false);
        acc[mt][nt] = __builtin_amdgcn_wmma_f32_16x16x32_bf16(
            false, aL[mt], false, bH[nt], (short)0, acc[mt][nt], false, false);
      }
  };

  const int kIters = K / BK;
  stage_all(0, 0);
  // steady state: stage next buffer, wait for current, compute
  for (int kk = 0; kk < kIters - 1; ++kk) {
    const int cur = kk & 1;
    stage_all(cur ^ 1, (kk + 1) * BK);
#if USE_TDM
    // Non-issuing waves have TENSORcnt==0: wait retires immediately (no branch).
    __builtin_amdgcn_s_wait_tensorcnt(1);
#else
    asm volatile("s_wait_asynccnt 0x6" ::: "memory");
#endif
    __syncthreads();
    compute_buf(cur);
    __syncthreads();
  }
  // last iteration (no prefetch)
#if USE_TDM
  __builtin_amdgcn_s_wait_tensorcnt(0);
#else
  asm volatile("s_wait_asynccnt 0x0" ::: "memory");
#endif
  __syncthreads();
  compute_buf((kIters - 1) & 1);

  // Epilogue: C layout lane = column (lane%16), VGPR rr = row rr + (lane/16)*8
#pragma unroll
  for (int mt = 0; mt < 2; ++mt)
#pragma unroll
    for (int nt = 0; nt < 2; ++nt) {
      int col = n0 + wn * 32 + nt * 16 + r;
      float bv = bias[col];
      int rowBase = m0 + wm * 32 + mt * 16 + half * 8;
#pragma unroll
      for (int rr = 0; rr < 8; ++rr)
        C[(size_t)(rowBase + rr) * N + col] = acc[mt][nt][rr] + bv;
    }
}

// ---------- elementwise gates + next-h bf16 split ----------
__global__ __launch_bounds__(256) void lstm_gates_kernel(
    const float* __restrict__ xproj, const float* __restrict__ hproj,
    const float* __restrict__ cin, float* __restrict__ cout,
    float* __restrict__ hout,  // may be null (non-final steps)
    unsigned short* __restrict__ hhi, unsigned short* __restrict__ hlo) {
  int j = blockIdx.x * 256 + threadIdx.x;  // j in [0, B*H)
  int b = j >> 10;                         // / kH
  int n = j & (kH - 1);
  size_t ro = (size_t)b * kG;
  float pf = xproj[ro + n]          + hproj[ro + n];
  float pi = xproj[ro + kH + n]     + hproj[ro + kH + n];
  float po = xproj[ro + 2 * kH + n] + hproj[ro + 2 * kH + n];
  float pg = xproj[ro + 3 * kH + n] + hproj[ro + 3 * kH + n];
  float f  = 1.0f / (1.0f + __expf(-pf));
  float i_ = 1.0f / (1.0f + __expf(-pi));
  float o_ = 1.0f / (1.0f + __expf(-po));
  float g  = tanhf(pg);
  float cn = f * cin[j] + i_ * g;
  float hn = o_ * tanhf(cn);
  cout[j] = cn;
  if (hout) hout[j] = hn;
  unsigned short hb = f2bf(hn);
  hhi[j] = hb;
  hlo[j] = f2bf(hn - bf2f(hb));
}

extern "C" void kernel_launch(void* const* d_in, const int* in_sizes, int n_in,
                              void* d_out, int out_size, void* d_ws, size_t ws_size,
                              hipStream_t stream) {
  (void)in_sizes; (void)n_in; (void)out_size; (void)ws_size;
  const float* inputs = (const float*)d_in[0];
  const float* W      = (const float*)d_in[1];
  const float* bW     = (const float*)d_in[2];
  const float* U      = (const float*)d_in[3];
  const float* bU     = (const float*)d_in[4];
  // d_in[5] = num_layers (device scalar); fixed-config kernel uses kL = 8.

  const size_t PROJ  = (size_t)kB * kG * sizeof(float);           // 64 MB
  const size_t HALF  = (size_t)kG * kIn * sizeof(unsigned short); // 8 MB (= B*H too)
  const size_t CBUF  = (size_t)kB * kH * sizeof(float);           // 16 MB

  char* ws = (char*)d_ws;
  size_t off = 0;
  auto take = [&](size_t bytes) { char* p = ws + off; off += bytes; return p; };
  float*          xproj = (float*)take(PROJ);
  float*          hproj = (float*)take(PROJ);
  unsigned short* Whi   = (unsigned short*)take(HALF);
  unsigned short* Wlo   = (unsigned short*)take(HALF);
  unsigned short* Uhi   = (unsigned short*)take(HALF);
  unsigned short* Ulo   = (unsigned short*)take(HALF);
  unsigned short* inhi  = (unsigned short*)take(HALF);
  unsigned short* inlo  = (unsigned short*)take(HALF);
  unsigned short* hhi   = (unsigned short*)take(HALF);
  unsigned short* hlo   = (unsigned short*)take(HALF);
  float*          cbuf  = (float*)take(CBUF);

  const int nSplit = kG * kIn;                 // 4,194,304 (same for all three)
  const int splitBlocks = nSplit / 256;
  lstm_split_kernel<<<splitBlocks, 256, 0, stream>>>(W, Whi, Wlo, nSplit);
  lstm_split_kernel<<<splitBlocks, 256, 0, stream>>>(U, Uhi, Ulo, nSplit);
  lstm_split_kernel<<<splitBlocks, 256, 0, stream>>>(inputs, inhi, inlo, nSplit);

  hipMemsetAsync(hhi, 0, HALF, stream);        // h0 = 0 (bf16 zero bits)
  hipMemsetAsync(hlo, 0, HALF, stream);
  hipMemsetAsync(cbuf, 0, CBUF, stream);       // c0 = 0

  dim3 ggrid(kG / BN, kB / BM);                // (64, 32)
  lstm_gemm_nt<<<ggrid, 256, 0, stream>>>(inhi, inlo, Whi, Wlo, bW, xproj,
                                          kB, kG, kIn);

  float* hOutFinal = (float*)d_out;                    // ht
  float* cOutFinal = (float*)d_out + (size_t)kB * kH;  // ct
  const int gateBlocks = (kB * kH) / 256;              // 16384
  for (int s = 0; s < kL; ++s) {
    lstm_gemm_nt<<<ggrid, 256, 0, stream>>>(hhi, hlo, Uhi, Ulo, bU, hproj,
                                            kB, kG, kH);
    const bool last = (s == kL - 1);
    lstm_gates_kernel<<<gateBlocks, 256, 0, stream>>>(
        xproj, hproj, cbuf,
        last ? cOutFinal : cbuf,
        last ? hOutFinal : nullptr,
        hhi, hlo);
  }
}
